// GruDecoder_57535381897874
// MI455X (gfx1250) — matrix operand order repaired and checked by reading, not compiled, not run on hardware
//
#include <hip/hip_runtime.h>
#include <hip/hip_bf16.h>
#include <math.h>

// ---------------------------------------------------------------------------
// GRU decoder (teacher forcing) for MI455X / gfx1250, wave32 + WMMA bf16.
//   V=32000 E=512 H=512 T=128 B=32. Dominant cost: [4096 x 32000 x 512] GEMM
//   (134 GFLOP, 524 MB output). bf16 WMMA (f32 accum) + flash-style logsoftmax.
//   K3 stages A-tiles into LDS via global_load_async_to_lds_b128 (ASYNCcnt).
// ---------------------------------------------------------------------------

#define VV 32000
#define EE 512
#define HH 512
#define TT 128
#define BB 32
#define H3 1536            // 3*H
#define MR 4096            // B*T rows
#define NCHUNK 250         // 32000 / 128

typedef __bf16 v16bf __attribute__((ext_vector_type(16)));
typedef float  v8f   __attribute__((ext_vector_type(8)));

union Frag { v16bf v; unsigned u[8]; };

__device__ __forceinline__ unsigned short f2bf(float f) {
  union { float f; unsigned u; } x; x.f = f;
  unsigned r = x.u + 0x7FFFu + ((x.u >> 16) & 1u);   // round-to-nearest-even
  return (unsigned short)(r >> 16);
}
__device__ __forceinline__ unsigned pack2bf(float a, float b) {
  return (unsigned)f2bf(a) | ((unsigned)f2bf(b) << 16);
}
__device__ __forceinline__ float bf2f(unsigned short h) {
  union { unsigned u; float f; } x; x.u = ((unsigned)h) << 16; return x.f;
}
__device__ __forceinline__ float sigmoidf_(float x) {
  return 1.0f / (1.0f + __expf(-x));
}

#define WMMA_BF16(A, Bf, C) \
  __builtin_amdgcn_wmma_f32_16x16x32_bf16(false, (A).v, false, (Bf).v, (short)0, (C), false, false)

// ---------------------------------------------------------------------------
// K0: transpose + bf16-pair-pack weights: W[N][K] f32 -> out[K/2][N] u32,
// out[(k>>1)*N + n] = pack(bf16 W[n][k], bf16 W[n][k+1]).  WMMA B-fragment
// register v then becomes a single coalesced dword load per lane.
// ---------------------------------------------------------------------------
__global__ void k_pack_wT(const float* __restrict__ W, unsigned* __restrict__ out,
                          int N, int K) {
  long long idx = (long long)blockIdx.x * blockDim.x + threadIdx.x;
  long long total = (long long)N * (K >> 1);
  if (idx >= total) return;
  int n  = (int)(idx % N);
  int kh = (int)(idx / N);
  const float* p = W + (size_t)n * K + 2 * kh;
  out[idx] = pack2bf(p[0], p[1]);
}

// ---------------------------------------------------------------------------
// K1: gi = embedding[ids] @ W_ih^T + b_ih  -> gi[T*B][3H] f32 (row = t*B+b).
// 4 waves/WG, each wave one 16x16 tile; WG tile = 16 rows x 64 cols.
// A: embedding rows gathered per lane, converted f32->bf16 in-register.
// ---------------------------------------------------------------------------
__global__ __launch_bounds__(128)
void k_gi_gemm(const float* __restrict__ emb,        // [V][E] f32
               const int* __restrict__ tgt,          // [B][T] i32
               const unsigned* __restrict__ WihTp,   // [E/2][3H] packed bf16
               const float* __restrict__ b_ih,       // [3H]
               float* __restrict__ gi)               // [T*B][3H]
{
  const int lane = threadIdx.x & 31;
  const int wv   = threadIdx.x >> 5;        // 0..3
  const int l15  = lane & 15;
  const int hi8  = (lane >> 4) * 8;
  const int rt   = blockIdx.y;              // row tile: 0..255
  const int n    = blockIdx.x * 64 + wv * 16 + l15;   // 0..1535

  const int mrow = rt * 16 + l15;
  const int t = mrow >> 5, b = mrow & 31;
  const int id = (t == 0) ? 1 : tgt[b * TT + (t - 1)];   // SOS_TOKEN_ID = 1
  const float* arow = emb + (size_t)id * EE;

  v8f acc = {};
  for (int kc = 0; kc < 16; ++kc) {        // K = 512 = 16 chunks of 32
    const int kbase = kc * 32;
    Frag a, bf;
    const int ko = kbase + 2 * hi8;
#pragma unroll
    for (int j = 0; j < 4; ++j) {
      a.u[j]     = pack2bf(arow[ko + 2*j],      arow[ko + 2*j + 1]);
      a.u[4 + j] = pack2bf(arow[ko + 8 + 2*j], arow[ko + 8 + 2*j + 1]);
    }
    const int rb = (kbase >> 1) + hi8;
#pragma unroll
    for (int v = 0; v < 8; ++v)
      bf.u[v] = WihTp[(size_t)(rb + v) * H3 + n];
    acc = WMMA_BF16(a, bf, acc);
  }
  const float bias = b_ih[n];
#pragma unroll
  for (int i = 0; i < 8; ++i) {
    const int m = rt * 16 + i + hi8;
    gi[(size_t)m * H3 + n] = acc[i] + bias;
  }
}

// ---------------------------------------------------------------------------
// K2: persistent GRU recurrence. One workgroup (32 waves = 1 WGP), hidden
// state ping-ponged in LDS as bf16 (2 x 32KB). Wave w owns hidden col tile
// jt = w and computes BOTH batch halves against one set of 3 gate B-frags
// (halves W_hh fragment traffic). __syncthreads() is the per-step barrier.
// ---------------------------------------------------------------------------
__global__ __launch_bounds__(1024)
void k_gru_scan(const float* __restrict__ h0,        // [32][512] f32
                const float* __restrict__ gi,        // [128][32][1536]
                const unsigned* __restrict__ WhhTp,  // [256][1536] packed bf16
                const float* __restrict__ b_hh,      // [1536]
                unsigned short* __restrict__ hs)     // [128*32][512] bf16
{
  __shared__ unsigned short hbuf[2][BB * HH];        // 64 KB ping-pong
  const int tid = threadIdx.x;
  for (int i = tid; i < BB * HH; i += 1024) hbuf[0][i] = f2bf(h0[i]);
  __syncthreads();

  const int lane = tid & 31, jt = tid >> 5;          // wave = j-tile 0..31
  const int l15 = lane & 15, hi8 = (lane >> 4) * 8;
  const int jr = jt * 16 + l15;

  for (int t = 0; t < TT; ++t) {
    const int cur = t & 1, nxt = cur ^ 1;
    const float* git = gi + (size_t)t * BB * H3;
    v8f a_r[2] = {{}, {}}, a_z[2] = {{}, {}}, a_n[2] = {{}, {}};
    for (int kc = 0; kc < 16; ++kc) {
      const int kbase = kc * 32;
      Frag br, bz, bn;
      const int rb = (kbase >> 1) + hi8;
#pragma unroll
      for (int v = 0; v < 8; ++v) {
        const unsigned* wr = WhhTp + (size_t)(rb + v) * H3 + jr;
        br.u[v] = wr[0];
        bz.u[v] = wr[HH];
        bn.u[v] = wr[2 * HH];
      }
#pragma unroll
      for (int mt = 0; mt < 2; ++mt) {
        Frag a;
        const unsigned short* hp = &hbuf[cur][(mt * 16 + l15) * HH + kbase + 2 * hi8];
#pragma unroll
        for (int j = 0; j < 4; ++j) {
          a.u[j]     = *(const unsigned*)(hp + 2 * j);
          a.u[4 + j] = *(const unsigned*)(hp + 8 + 2 * j);
        }
        a_r[mt] = WMMA_BF16(a, br, a_r[mt]);
        a_z[mt] = WMMA_BF16(a, bz, a_z[mt]);
        a_n[mt] = WMMA_BF16(a, bn, a_n[mt]);
      }
    }
#pragma unroll
    for (int mt = 0; mt < 2; ++mt) {
#pragma unroll
      for (int i = 0; i < 8; ++i) {
        const int brow = mt * 16 + i + hi8;
        const int j    = jt * 16 + l15;
        const float* g = git + (size_t)brow * H3 + j;
        const float rr = a_r[mt][i] + b_hh[j]          + g[0];
        const float zz = a_z[mt][i] + b_hh[HH + j]     + g[HH];
        const float nh = a_n[mt][i] + b_hh[2 * HH + j];
        const float ni = g[2 * HH];
        const float r = sigmoidf_(rr);
        const float z = sigmoidf_(zz);
        const float hprev = bf2f(hbuf[cur][brow * HH + j]);
        const float nn = tanhf(ni + r * nh);
        const float h  = (1.0f - z) * nn + z * hprev;
        const unsigned short hb = f2bf(h);
        hbuf[nxt][brow * HH + j] = hb;
        hs[((size_t)t * BB + brow) * HH + j] = hb;
      }
    }
    __syncthreads();     // step t: hbuf[nxt] published; WAR on hbuf[cur] safe
  }
}

// ---------------------------------------------------------------------------
// K3: logits = hs @ W_out^T + b_out -> d_out (raw), plus per-(row, 128-col
// chunk) flash-softmax partials. WG = 128 thr (4 waves); WG tile = 64 rows x
// 128 cols; each wave: 4 M-tiles x 2 N-tiles (8 WMMA per B-fragment pair).
// A tile (64x32 bf16) double-buffered in LDS via async global->LDS B128 DMA.
// LDS row stride = 40 elems (80 B): 16B-aligned and conflict-free frag reads.
// ---------------------------------------------------------------------------
__device__ __forceinline__ void stageA_async(unsigned short (*sAb)[40],
                                             const unsigned short* hsbase,
                                             int kbase, int tid)
{
#pragma unroll
  for (int q = 0; q < 2; ++q) {
    const int idx = tid + 128 * q;          // 0..255 -> 64 rows x 4 segs of 16B
    const int row = idx >> 2, seg = idx & 3;
    const unsigned lds = (unsigned)(size_t)&sAb[row][seg * 8];
    const unsigned long long ga =
        (unsigned long long)(const void*)(hsbase + (size_t)row * HH + kbase + seg * 8);
    asm volatile("global_load_async_to_lds_b128 %0, %1, off"
                 :: "v"(lds), "v"(ga) : "memory");
  }
}
__device__ __forceinline__ void wait_async_barrier() {
  asm volatile("s_wait_asynccnt 0x0" ::: "memory");
  __syncthreads();
}

__global__ __launch_bounds__(128)
void k_out_gemm(const unsigned short* __restrict__ hs,  // [4096][512] bf16
                const unsigned* __restrict__ WoutTp,    // [256][32000] packed bf16
                const float* __restrict__ b_out,        // [32000]
                float* __restrict__ out,                // [B][T][V]
                float* __restrict__ pMax,               // [4096][250]
                float* __restrict__ pSum)               // [4096][250]
{
  __shared__ unsigned short sA[2][64][40];              // 10 KB, double-buffered
  __shared__ float sM[4][64], sS[4][64];
  const int tid = threadIdx.x;
  const int lane = tid & 31, wv = tid >> 5;
  const int l15 = lane & 15, hi8 = (lane >> 4) * 8;
  const int rg    = blockIdx.y;               // 64-row group: 0..63
  const int chunk = blockIdx.x;               // 0..249
  const unsigned short* hsbase = hs + (size_t)rg * 64 * HH;
  const int n0 = chunk * 128 + wv * 32 + l15;

  v8f acc0[4] = {{}, {}, {}, {}};             // N-tile n0
  v8f acc1[4] = {{}, {}, {}, {}};             // N-tile n0+16

  stageA_async(sA[0], hsbase, 0, tid);
  wait_async_barrier();

  for (int kc = 0; kc < 16; ++kc) {
    const int buf = kc & 1;
    if (kc < 15) stageA_async(sA[buf ^ 1], hsbase, (kc + 1) * 32, tid);

    const int rb = (kc * 16) + hi8;           // (kbase>>1) + hi8
    Frag b0, b1;
#pragma unroll
    for (int v = 0; v < 8; ++v) {
      const unsigned* wr = WoutTp + (size_t)(rb + v) * VV + n0;
      b0.u[v] = wr[0];
      b1.u[v] = wr[16];
    }
#pragma unroll
    for (int q = 0; q < 4; ++q) {
      Frag a;
      const unsigned short* hp = &sA[buf][q * 16 + l15][2 * hi8];
#pragma unroll
      for (int j = 0; j < 4; ++j) {
        a.u[j]     = *(const unsigned*)(hp + 2 * j);
        a.u[4 + j] = *(const unsigned*)(hp + 8 + 2 * j);
      }
      acc0[q] = WMMA_BF16(a, b0, acc0[q]);
      acc1[q] = WMMA_BF16(a, b1, acc1[q]);
    }
    wait_async_barrier();                     // publish buf^1; all done with buf
  }

  const float bias0 = b_out[n0], bias1 = b_out[n0 + 16];
#pragma unroll
  for (int q = 0; q < 4; ++q) {
    float mx[8], se[8];
#pragma unroll
    for (int i = 0; i < 8; ++i) {
      const int row = rg * 64 + q * 16 + i + hi8;
      const int b = row & 31, t = row >> 5;
      const float v0 = acc0[q][i] + bias0;
      const float v1 = acc1[q][i] + bias1;
      const size_t obase = ((size_t)b * TT + t) * VV;
      out[obase + n0]      = v0;
      out[obase + n0 + 16] = v1;
      float m = fmaxf(v0, v1);
#pragma unroll
      for (int s = 1; s < 16; s <<= 1) m = fmaxf(m, __shfl_xor(m, s, 32));
      float e = __expf(v0 - m) + __expf(v1 - m);
#pragma unroll
      for (int s = 1; s < 16; s <<= 1) e += __shfl_xor(e, s, 32);
      mx[i] = m; se[i] = e;
    }
    if (l15 == 0) {
#pragma unroll
      for (int i = 0; i < 8; ++i) {
        sM[wv][q * 16 + i + hi8] = mx[i];
        sS[wv][q * 16 + i + hi8] = se[i];
      }
    }
  }
  __syncthreads();
  if (tid < 64) {
    const int r = tid;
    float M = sM[0][r];
    for (int w = 1; w < 4; ++w) M = fmaxf(M, sM[w][r]);
    float S = 0.0f;
    for (int w = 0; w < 4; ++w) S += sS[w][r] * __expf(sM[w][r] - M);
    const int row = rg * 64 + r;
    pMax[(size_t)row * NCHUNK + chunk] = M;
    pSum[(size_t)row * NCHUNK + chunk] = S;
  }
}

// ---------------------------------------------------------------------------
// K4: combine 250 partials per row -> rowOff = max + log(sum exp).
// ---------------------------------------------------------------------------
__global__ __launch_bounds__(256)
void k_row_lse(const float* __restrict__ pMax, const float* __restrict__ pSum,
               float* __restrict__ rowOff)
{
  __shared__ float sm[256], ss[256];
  const int row = blockIdx.x, tid = threadIdx.x;
  const float myM = (tid < NCHUNK) ? pMax[(size_t)row * NCHUNK + tid] : -3.4e38f;
  sm[tid] = myM;
  __syncthreads();
  for (int s = 128; s > 0; s >>= 1) {
    if (tid < s) sm[tid] = fmaxf(sm[tid], sm[tid + s]);
    __syncthreads();
  }
  const float M = sm[0];
  __syncthreads();
  ss[tid] = (tid < NCHUNK) ? pSum[(size_t)row * NCHUNK + tid] * __expf(myM - M) : 0.0f;
  __syncthreads();
  for (int s = 128; s > 0; s >>= 1) {
    if (tid < s) ss[tid] += ss[tid + s];
    __syncthreads();
  }
  if (tid == 0) rowOff[row] = M + logf(ss[0]);
}

// ---------------------------------------------------------------------------
// K5: out[b][t][v] -= rowOff[t*32+b], float4 vectorized (V % 4 == 0).
// ---------------------------------------------------------------------------
__global__ __launch_bounds__(256)
void k_normalize(float* __restrict__ out, const float* __restrict__ rowOff)
{
  const size_t idx = (size_t)blockIdx.x * blockDim.x + threadIdx.x;
  const size_t total4 = (size_t)BB * TT * VV / 4;      // 32,768,000
  if (idx >= total4) return;
  const size_t e = idx * 4;
  const int b = (int)(e / ((size_t)TT * VV));
  const size_t rem = e - (size_t)b * TT * VV;
  const int t = (int)(rem / VV);
  const float off = rowOff[t * BB + b];
  float4* p = (float4*)out + idx;
  float4 v = *p;
  v.x -= off; v.y -= off; v.z -= off; v.w -= off;
  *p = v;
}

// ---------------------------------------------------------------------------
extern "C" void kernel_launch(void* const* d_in, const int* in_sizes, int n_in,
                              void* d_out, int out_size, void* d_ws, size_t ws_size,
                              hipStream_t stream) {
  (void)in_sizes; (void)n_in; (void)out_size; (void)ws_size;
  // setup_inputs order:
  // 0 encoder_output (unused), 1 encoder_hidden_state, 2 tgt_seqs,
  // 3 tgt_seq_lengths (unused), 4 embedding, 5 W_ih, 6 W_hh, 7 b_ih,
  // 8 b_hh, 9 W_out, 10 b_out
  const float* h0    = (const float*)d_in[1];
  const int*   tgt   = (const int*)d_in[2];
  const float* emb   = (const float*)d_in[4];
  const float* W_ih  = (const float*)d_in[5];
  const float* W_hh  = (const float*)d_in[6];
  const float* b_ih  = (const float*)d_in[7];
  const float* b_hh  = (const float*)d_in[8];
  const float* W_out = (const float*)d_in[9];
  const float* b_out = (const float*)d_in[10];
  float* out = (float*)d_out;

  // Workspace carve-up (256B aligned), ~73.5 MB total.
  char* ws = (char*)d_ws;
  size_t off = 0;
  auto take = [&](size_t bytes) -> char* {
    char* p = ws + off;
    off = (off + bytes + 255) & ~(size_t)255;
    return p;
  };
  unsigned*       WihTp  = (unsigned*)      take((size_t)(EE/2) * H3 * 4);   // 1.5 MB
  unsigned*       WhhTp  = (unsigned*)      take((size_t)(HH/2) * H3 * 4);   // 1.5 MB
  unsigned*       WoutTp = (unsigned*)      take((size_t)(HH/2) * VV * 4);   // 32.8 MB
  float*          gi     = (float*)         take((size_t)MR * H3 * 4);       // 25.2 MB
  unsigned short* hs     = (unsigned short*)take((size_t)MR * HH * 2);       // 4.2 MB
  float*          pMax   = (float*)         take((size_t)MR * NCHUNK * 4);   // 4.1 MB
  float*          pSum   = (float*)         take((size_t)MR * NCHUNK * 4);   // 4.1 MB
  float*          rowOff = (float*)         take((size_t)MR * 4);

  // K0: pack/transpose weights to bf16 [K/2][N]
  {
    int tot_ih = H3 * (EE / 2);
    k_pack_wT<<<(tot_ih + 255) / 256, 256, 0, stream>>>(W_ih, WihTp, H3, EE);
    k_pack_wT<<<(tot_ih + 255) / 256, 256, 0, stream>>>(W_hh, WhhTp, H3, HH);
    int tot_out = VV * (HH / 2);
    k_pack_wT<<<(tot_out + 255) / 256, 256, 0, stream>>>(W_out, WoutTp, VV, HH);
  }
  // K1: input-side gate projections (embedding gather fused into GEMM A)
  k_gi_gemm<<<dim3(H3 / 64, MR / 16), 128, 0, stream>>>(emb, tgt, WihTp, b_ih, gi);
  // K2: persistent GRU scan (1 WGP, hidden state in LDS)
  k_gru_scan<<<1, 1024, 0, stream>>>(h0, gi, WhhTp, b_hh, hs);
  // K3: vocab projection + flash-softmax partials (async LDS staging)
  k_out_gemm<<<dim3(NCHUNK, MR / 64), 128, 0, stream>>>(hs, WoutTp, b_out, out, pMax, pSum);
  // K4: per-row log-sum-exp combine
  k_row_lse<<<MR, 256, 0, stream>>>(pMax, pSum, rowOff);
  // K5: normalize logits in place
  {
    size_t total4 = (size_t)BB * TT * VV / 4;
    k_normalize<<<(unsigned)((total4 + 255) / 256), 256, 0, stream>>>(out, rowOff);
  }
}
